// NeuralMemory_52742198395035
// MI455X (gfx1250) — matrix operand order repaired and checked by reading, not compiled, not run on hardware
//
#include <hip/hip_runtime.h>
#include <hip/hip_bf16.h>
#include <math.h>

typedef __attribute__((ext_vector_type(16))) __bf16 v16bf;
typedef __attribute__((ext_vector_type(8)))  float  v8f;

#define WMMA_BF16(acc, a, b) \
  (acc) = __builtin_amdgcn_wmma_f32_16x16x32_bf16(false, (a), false, (b), (short)0, (acc), false, false)

// ---------------- math helpers ----------------
__device__ __forceinline__ float sigf(float x)   { return 1.f / (1.f + __expf(-x)); }
__device__ __forceinline__ float siluf(float x)  { return x * sigf(x); }
__device__ __forceinline__ float dsiluf(float x) { float s = sigf(x); return s * (1.f + x * (1.f - s)); }

// ---------------- WMMA fragment builders (contiguous k order k = 16*(L>>4) + element, wave32) ----
// A 16x32 (MxK) bf16: lane L: m=L&15; lanes>=16 add K+16; VGPR v: K=2v,2v+1
__device__ __forceinline__ v16bf frag_a_bf(const __bf16* s, int ld, int m0, int k0) {
  int lane = threadIdx.x & 31;
  int m = m0 + (lane & 15), kh = (lane >> 4) * 16;
  v16bf a;
#pragma unroll
  for (int v = 0; v < 8; ++v) {
    int kb = 2 * v + kh + k0;
    a[2 * v]     = s[m * ld + kb];
    a[2 * v + 1] = s[m * ld + kb + 1];
  }
  return a;
}
__device__ __forceinline__ v16bf frag_a_f32(const float* s, int ld, int m0, int k0) {
  int lane = threadIdx.x & 31;
  int m = m0 + (lane & 15), kh = (lane >> 4) * 16;
  v16bf a;
#pragma unroll
  for (int v = 0; v < 8; ++v) {
    int kb = 2 * v + kh + k0;
    a[2 * v]     = (__bf16)s[m * ld + kb];
    a[2 * v + 1] = (__bf16)s[m * ld + kb + 1];
  }
  return a;
}
// A[m,k] = s[(k0+k)*ld + m0+m] * coef[k0+k]   (transposed f32 source, per-k scalar)
__device__ __forceinline__ v16bf frag_at_scaled(const float* s, int ld, int m0, int k0,
                                                const float* coef) {
  int lane = threadIdx.x & 31;
  int m = m0 + (lane & 15), kh = (lane >> 4) * 16;
  v16bf a;
#pragma unroll
  for (int v = 0; v < 8; ++v) {
    int k = 2 * v + kh + k0;
    a[2 * v]     = (__bf16)(s[k * ld + m] * coef[k]);
    a[2 * v + 1] = (__bf16)(s[(k + 1) * ld + m] * coef[k + 1]);
  }
  return a;
}
// B 32x16 (KxN): lane L: n=L&15; lanes>=16 add K+16; VGPR v: K=2v,2v+1
__device__ __forceinline__ v16bf frag_b_bf(const __bf16* s, int ld, int k0, int n0) {
  int lane = threadIdx.x & 31;
  int n = n0 + (lane & 15), ko = (lane >> 4) * 16 + k0;
  v16bf b;
#pragma unroll
  for (int v = 0; v < 8; ++v) {
    b[2 * v]     = s[(ko + 2 * v) * ld + n];
    b[2 * v + 1] = s[(ko + 2 * v + 1) * ld + n];
  }
  return b;
}
// B[k,n] = s[(n0+n)*ld + k0+k]   (implicit transpose of a row-major weight)
__device__ __forceinline__ v16bf frag_bt_bf(const __bf16* s, int ld, int k0, int n0) {
  int lane = threadIdx.x & 31;
  int n = n0 + (lane & 15), ko = (lane >> 4) * 16 + k0;
  v16bf b;
#pragma unroll
  for (int v = 0; v < 8; ++v) {
    b[2 * v]     = s[n * ld + ko + 2 * v];
    b[2 * v + 1] = s[n * ld + ko + 2 * v + 1];
  }
  return b;
}
// B[k,n] = silu(src_f32[(k0+k)*ld + n0+n])
__device__ __forceinline__ v16bf frag_b_silu(const float* s, int ld, int k0, int n0) {
  int lane = threadIdx.x & 31;
  int n = n0 + (lane & 15), ko = (lane >> 4) * 16 + k0;
  v16bf b;
#pragma unroll
  for (int v = 0; v < 8; ++v) {
    b[2 * v]     = (__bf16)siluf(s[(ko + 2 * v) * ld + n]);
    b[2 * v + 1] = (__bf16)siluf(s[(ko + 2 * v + 1) * ld + n]);
  }
  return b;
}

// =====================================================================
// Kernel 1: k/v/q projections (y = x @ W^T), + per-head L2 normalize k,q
//   grid: 3 * 64 * 4 blocks of 256 threads; 64x64 output tile per block
// =====================================================================
__global__ __launch_bounds__(256) void proj_kernel(
    const float* __restrict__ x, const float* __restrict__ Wk,
    const float* __restrict__ Wv, const float* __restrict__ Wq,
    float* __restrict__ kbuf, float* __restrict__ vbuf, float* __restrict__ qbuf) {
  extern __shared__ char smem[];
  __bf16* xh = (__bf16*)smem;          // 64 x 256
  __bf16* wh = xh + 64 * 256;          // 64 x 256 (rows = out features)
  float*  yt = (float*)(wh + 64 * 256);// 64 x 64
  float*  rs = yt + 64 * 64;           // 64 row scales

  int bx   = blockIdx.x;
  int widx = bx >> 8;                  // 0:k 1:v 2:q
  int rem  = bx & 255;
  int m0   = (rem >> 2) * 64;
  int n0   = (rem & 3) * 64;
  const float* W    = (widx == 0) ? Wk : (widx == 1) ? Wv : Wq;
  float*       outb = (widx == 0) ? kbuf : (widx == 1) ? vbuf : qbuf;
  int tid = threadIdx.x, wv = tid >> 5, lane = tid & 31;

  for (int i = tid; i < 64 * 256; i += 256) {
    int r = i >> 8, c = i & 255;
    xh[i] = (__bf16)x[(m0 + r) * 256 + c];
    wh[i] = (__bf16)W[(n0 + r) * 256 + c];
  }
  __syncthreads();

  for (int t = 0; t < 2; ++t) {
    int tt = wv + t * 8;
    int mr = (tt >> 2) * 16, nc = (tt & 3) * 16;
    v8f acc = {};
#pragma unroll
    for (int kk = 0; kk < 256; kk += 32) {
      v16bf a = frag_a_bf(xh, 256, mr, kk);
      v16bf b = frag_bt_bf(wh, 256, kk, nc);
      WMMA_BF16(acc, a, b);
    }
#pragma unroll
    for (int v = 0; v < 8; ++v) {
      int m = mr + v + ((lane >> 4) * 8), n = nc + (lane & 15);
      yt[m * 64 + n] = acc[v];
    }
  }
  __syncthreads();

  if (widx != 1) { // 64-col tile == exactly one head (HD=64): normalize rows
    if (tid < 64) {
      float s2 = 0.f;
      for (int j = 0; j < 64; ++j) { float y = yt[tid * 64 + j]; s2 += y * y; }
      rs[tid] = 1.f / fmaxf(sqrtf(s2), 1e-12f);
    }
    __syncthreads();
    for (int i = tid; i < 64 * 64; i += 256) {
      int r = i >> 6, c = i & 63;
      outb[(m0 + r) * 256 + n0 + c] = yt[i] * rs[r];
    }
  } else {
    for (int i = tid; i < 64 * 64; i += 256) {
      int r = i >> 6, c = i & 63;
      outb[(m0 + r) * 256 + n0 + c] = yt[i];
    }
  }
}

// =====================================================================
// Kernel 2: gates lr/eta/alpha = sigmoid(x @ Wg^T)  (tiny: plain VALU)
// =====================================================================
__global__ __launch_bounds__(256) void gates_kernel(
    const float* __restrict__ x, const float* __restrict__ Wlr,
    const float* __restrict__ Wmom, const float* __restrict__ Wdec,
    float* __restrict__ lrb, float* __restrict__ etab, float* __restrict__ alb) {
  int id = blockIdx.x * 256 + threadIdx.x;
  if (id >= 4096 * 12) return;
  int tok = id / 12, r = id % 12, h = r / 3, g = r % 3;
  const float* W = (g == 0) ? Wlr : (g == 1) ? Wmom : Wdec;
  float acc = 0.f;
  for (int i = 0; i < 256; ++i) acc += x[tok * 256 + i] * W[h * 256 + i];
  float val = sigf(acc);
  float* ob = (g == 0) ? lrb : (g == 1) ? etab : alb;
  ob[tok * 4 + h] = val;
}

// =====================================================================
// Kernel 3: per-(b,h) neural memory. 16 blocks x 256 thr (8 wave32).
//   State (W1,W2,M1,M2) fp32 resident in LDS across the whole sequence.
//   Token scan collapsed to closed-form scalar coefficients + rank-1-sum
//   GEMMs; grad clip via ||a x b||_F = ||a||*||b||.
// =====================================================================
__global__ __launch_bounds__(256) void memory_kernel(
    const float* __restrict__ kbuf, const float* __restrict__ vbuf,
    const float* __restrict__ qbuf, const float* __restrict__ lrb,
    const float* __restrict__ etab, const float* __restrict__ alb,
    const float* __restrict__ W1g, const float* __restrict__ W2g,
    float* __restrict__ hob) {
  extern __shared__ char smem[];
  float* W1f  = (float*)smem;     // 128x64
  float* W2f  = W1f + 8192;       // 64x128
  float* M1f  = W2f + 8192;
  float* M2f  = M1f + 8192;
  float* h1f  = M2f + 8192;       // 64x128 pre-activations (k path)
  float* dh1f = h1f + 8192;       // 64x128 (includes the factor 2)
  float* rf   = dh1f + 8192;      // 64x64 residual
  __bf16* W1h = (__bf16*)(rf + 4096);
  __bf16* W2h = W1h + 8192;
  __bf16* qh  = W2h + 8192;       // 64x64
  __bf16* kh  = qh + 4096;
  __bf16* sb  = kh + 4096;        // 64x128 silu buffer
  float* lrA  = (float*)(sb + 8192);
  float* etA  = lrA + 64;  float* alA  = etA + 64;
  float* c1   = alA + 64;  float* c2   = c1 + 64;
  float* Esuf = c2 + 64;   float* Bsuf = Esuf + 64; float* Ssuf = Bsuf + 64;
  float* cu1  = Ssuf + 64; float* cw1  = cu1 + 64;
  float* cu2  = cw1 + 64;  float* cw2  = cu2 + 64;
  float* scal = cw2 + 64;  // [0]=E [1]=F [2]=G

  int b = blockIdx.x >> 2, h = blockIdx.x & 3;
  int tid = threadIdx.x, wv = tid >> 5, lane = tid & 31;
  const int rowbase = b * 1024;

  for (int i = tid; i < 8192; i += 256) {
    W1f[i] = W1g[h * 8192 + i];
    W2f[i] = W2g[h * 8192 + i];
    M1f[i] = 0.f; M2f[i] = 0.f;
  }
  __syncthreads();

  for (int c = 0; c < 16; ++c) {
    int s0 = c * 64;
    // ---- Phase A: bf16 weight copies + chunk activations + gate scalars
    for (int i = tid; i < 8192; i += 256) {
      W1h[i] = (__bf16)W1f[i];
      W2h[i] = (__bf16)W2f[i];
    }
    for (int i = tid; i < 4096; i += 256) {
      int t = i >> 6, d = i & 63;
      int g = (rowbase + s0 + t) * 256 + h * 64 + d;
      qh[i] = (__bf16)qbuf[g];
      kh[i] = (__bf16)kbuf[g];
    }
    if (tid < 64) {
      int gi = (rowbase + s0 + tid) * 4 + h;
      lrA[tid] = lrb[gi]; etA[tid] = etab[gi]; alA[tid] = alb[gi];
    }
    __syncthreads();

    // ---- Phase B1: hq = q @ W1^T ; silu -> sb   (64x128, K=64)
    for (int t = 0; t < 4; ++t) {
      int tt = wv + t * 8, m0 = (tt >> 3) * 16, j0 = (tt & 7) * 16;
      v8f acc = {};
#pragma unroll
      for (int kk = 0; kk < 64; kk += 32) {
        v16bf a = frag_a_bf(qh, 64, m0, kk);
        v16bf bf = frag_bt_bf(W1h, 64, kk, j0);
        WMMA_BF16(acc, a, bf);
      }
#pragma unroll
      for (int v = 0; v < 8; ++v) {
        int m = m0 + v + ((lane >> 4) * 8), n = j0 + (lane & 15);
        sb[m * 128 + n] = (__bf16)siluf(acc[v]);
      }
    }
    __syncthreads();
    // ---- Phase B2: out = q + silu(hq) @ W2^T -> hob  (64x64, K=128)
    for (int t = 0; t < 2; ++t) {
      int tt = wv + t * 8, m0 = (tt >> 2) * 16, d0 = (tt & 3) * 16;
      v8f acc = {};
#pragma unroll
      for (int kk = 0; kk < 128; kk += 32) {
        v16bf a = frag_a_bf(sb, 128, m0, kk);
        v16bf bf = frag_bt_bf(W2h, 128, kk, d0);
        WMMA_BF16(acc, a, bf);
      }
#pragma unroll
      for (int v = 0; v < 8; ++v) {
        int m = m0 + v + ((lane >> 4) * 8), n = d0 + (lane & 15);
        int g = (rowbase + s0 + m) * 256 + h * 64 + n;
        hob[g] = qbuf[g] + acc[v];
      }
    }
    __syncthreads();
    // ---- Phase C1: hk = k @ W1^T -> h1f ; silu -> sb
    for (int t = 0; t < 4; ++t) {
      int tt = wv + t * 8, m0 = (tt >> 3) * 16, j0 = (tt & 7) * 16;
      v8f acc = {};
#pragma unroll
      for (int kk = 0; kk < 64; kk += 32) {
        v16bf a = frag_a_bf(kh, 64, m0, kk);
        v16bf bf = frag_bt_bf(W1h, 64, kk, j0);
        WMMA_BF16(acc, a, bf);
      }
#pragma unroll
      for (int v = 0; v < 8; ++v) {
        int m = m0 + v + ((lane >> 4) * 8), n = j0 + (lane & 15);
        h1f[m * 128 + n] = acc[v];
        sb[m * 128 + n]  = (__bf16)siluf(acc[v]);
      }
    }
    __syncthreads();
    // ---- Phase C2: r = k + silu(hk) @ W2^T - v -> rf
    for (int t = 0; t < 2; ++t) {
      int tt = wv + t * 8, m0 = (tt >> 2) * 16, d0 = (tt & 3) * 16;
      v8f acc = {};
#pragma unroll
      for (int kk = 0; kk < 128; kk += 32) {
        v16bf a = frag_a_bf(sb, 128, m0, kk);
        v16bf bf = frag_bt_bf(W2h, 128, kk, d0);
        WMMA_BF16(acc, a, bf);
      }
#pragma unroll
      for (int v = 0; v < 8; ++v) {
        int m = m0 + v + ((lane >> 4) * 8), n = d0 + (lane & 15);
        int g = (rowbase + s0 + m) * 256 + h * 64 + n;
        rf[m * 64 + n] = kbuf[g] + acc[v] - vbuf[g];
      }
    }
    __syncthreads();
    // ---- Phase D: ds = 2 r @ W2 ; dh1 = ds * silu'(h1) -> dh1f
    for (int t = 0; t < 4; ++t) {
      int tt = wv + t * 8, m0 = (tt >> 3) * 16, j0 = (tt & 7) * 16;
      v8f acc = {};
#pragma unroll
      for (int kk = 0; kk < 64; kk += 32) {
        v16bf a = frag_a_f32(rf, 64, m0, kk);
        v16bf bf = frag_b_bf(W2h, 128, kk, j0);
        WMMA_BF16(acc, a, bf);
      }
#pragma unroll
      for (int v = 0; v < 8; ++v) {
        int m = m0 + v + ((lane >> 4) * 8), n = j0 + (lane & 15);
        int idx = m * 128 + n;
        dh1f[idx] = 2.f * acc[v] * dsiluf(h1f[idx]);
      }
    }
    __syncthreads();
    // ---- Phase E1: per-token grad-clip scales via rank-1 norm identity
    if (tid < 64) {
      float nr2 = 0.f, ns2 = 0.f, nd2 = 0.f, nk2 = 0.f;
      for (int d = 0; d < 64; ++d) {
        float r_ = rf[tid * 64 + d]; nr2 += r_ * r_;
        float kv = kbuf[(rowbase + s0 + tid) * 256 + h * 64 + d]; nk2 += kv * kv;
      }
      for (int j = 0; j < 128; ++j) {
        float sv = siluf(h1f[tid * 128 + j]); ns2 += sv * sv;
        float dv = dh1f[tid * 128 + j];       nd2 += dv * dv;
      }
      float g2 = 2.f * sqrtf(nr2 * ns2);   // ||2 r x s||_F
      float g1 = sqrtf(nd2 * nk2);         // ||dh1 x k||_F (2 already folded)
      c2[tid] = 1.f / fmaxf(g2 * 0.1f, 1.f);
      c1[tid] = 1.f / fmaxf(g1 * 0.1f, 1.f);
    }
    __syncthreads();
    // ---- Phase E2: closed-form scan coefficients (O(C), one lane)
    if (tid == 0) {
      float e = 1.f, a_ = 1.f;
      for (int t2 = 63; t2 >= 0; --t2) {
        Esuf[t2] = e;  Bsuf[t2] = a_;
        e *= etA[t2];  a_ *= (1.f - alA[t2]);
      }
      scal[0] = e;   // E = prod eta
      scal[1] = a_;  // F = prod (1-alpha)
      float S = 1.f; Ssuf[63] = 1.f;
      for (int t2 = 62; t2 >= 0; --t2) { S = Bsuf[t2] + etA[t2 + 1] * S; Ssuf[t2] = S; }
      float G = 0.f, ep = 1.f;
      for (int t2 = 0; t2 < 64; ++t2) { ep *= etA[t2]; G += Bsuf[t2] * ep; }
      scal[2] = G;
      for (int t2 = 0; t2 < 64; ++t2) {
        float base = -lrA[t2];
        cu1[t2] = base * c1[t2] * Esuf[t2];
        cw1[t2] = base * c1[t2] * Ssuf[t2];
        cu2[t2] = 2.f * base * c2[t2] * Esuf[t2];
        cw2[t2] = 2.f * base * c2[t2] * Ssuf[t2];
      }
    }
    __syncthreads();
    float Ee = scal[0], Ff = scal[1], Gg = scal[2];
    // ---- Phase F1: W1/M1 update: sum_t coef_t * dh1_t (x) k_t   (128x64, K=64)
    for (int t = 0; t < 4; ++t) {
      int tt = wv * 4 + t, j0 = (tt >> 2) * 16, d0 = (tt & 3) * 16;
      v8f aU = {}, aW = {};
#pragma unroll
      for (int kk = 0; kk < 64; kk += 32) {
        v16bf bf = frag_b_bf(kh, 64, kk, d0);
        v16bf au = frag_at_scaled(dh1f, 128, j0, kk, cu1);
        v16bf aw = frag_at_scaled(dh1f, 128, j0, kk, cw1);
        WMMA_BF16(aU, au, bf);
        WMMA_BF16(aW, aw, bf);
      }
#pragma unroll
      for (int v = 0; v < 8; ++v) {
        int m = j0 + v + ((lane >> 4) * 8), n = d0 + (lane & 15);
        int idx = m * 64 + n;
        float mo = M1f[idx], po = W1f[idx];
        M1f[idx] = Ee * mo + aU[v];
        W1f[idx] = Ff * po + Gg * mo + aW[v];
      }
    }
    // ---- Phase F2: W2/M2 update: sum_t coef_t * 2r_t (x) s_t   (64x128, K=64)
    for (int t = 0; t < 4; ++t) {
      int tt = wv * 4 + t, d0 = (tt >> 3) * 16, j0 = (tt & 7) * 16;
      v8f aU = {}, aW = {};
#pragma unroll
      for (int kk = 0; kk < 64; kk += 32) {
        v16bf bf = frag_b_silu(h1f, 128, kk, j0);
        v16bf au = frag_at_scaled(rf, 64, d0, kk, cu2);
        v16bf aw = frag_at_scaled(rf, 64, d0, kk, cw2);
        WMMA_BF16(aU, au, bf);
        WMMA_BF16(aW, aw, bf);
      }
#pragma unroll
      for (int v = 0; v < 8; ++v) {
        int m = d0 + v + ((lane >> 4) * 8), n = j0 + (lane & 15);
        int idx = m * 128 + n;
        float mo = M2f[idx], po = W2f[idx];
        M2f[idx] = Ee * mo + aU[v];
        W2f[idx] = Ff * po + Gg * mo + aW[v];
      }
    }
    __syncthreads();
  }
}

// =====================================================================
// Kernel 4: final projection out = headout @ Wout^T
// =====================================================================
__global__ __launch_bounds__(256) void outproj_kernel(
    const float* __restrict__ hob, const float* __restrict__ Wout,
    float* __restrict__ out) {
  extern __shared__ char smem[];
  __bf16* xh = (__bf16*)smem;
  __bf16* wh = xh + 64 * 256;
  int rem = blockIdx.x;
  int m0 = (rem >> 2) * 64, n0 = (rem & 3) * 64;
  int tid = threadIdx.x, wv = tid >> 5, lane = tid & 31;
  for (int i = tid; i < 64 * 256; i += 256) {
    int r = i >> 8, c = i & 255;
    xh[i] = (__bf16)hob[(m0 + r) * 256 + c];
    wh[i] = (__bf16)Wout[(n0 + r) * 256 + c];
  }
  __syncthreads();
  for (int t = 0; t < 2; ++t) {
    int tt = wv + t * 8;
    int mr = (tt >> 2) * 16, nc = (tt & 3) * 16;
    v8f acc = {};
#pragma unroll
    for (int kk = 0; kk < 256; kk += 32) {
      v16bf a = frag_a_bf(xh, 256, mr, kk);
      v16bf b = frag_bt_bf(wh, 256, kk, nc);
      WMMA_BF16(acc, a, b);
    }
#pragma unroll
    for (int v = 0; v < 8; ++v) {
      int m = mr + v + ((lane >> 4) * 8), n = nc + (lane & 15);
      out[(m0 + m) * 256 + n0 + n] = acc[v];
    }
  }
}

// =====================================================================
extern "C" void kernel_launch(void* const* d_in, const int* in_sizes, int n_in,
                              void* d_out, int out_size, void* d_ws, size_t ws_size,
                              hipStream_t stream) {
  (void)in_sizes; (void)n_in; (void)out_size; (void)ws_size;
  const float* x    = (const float*)d_in[0];
  const float* Wk   = (const float*)d_in[1];
  const float* Wv   = (const float*)d_in[2];
  const float* Wq   = (const float*)d_in[3];
  const float* Wlr  = (const float*)d_in[4];
  const float* Wmom = (const float*)d_in[5];
  const float* Wdec = (const float*)d_in[6];
  const float* W1   = (const float*)d_in[7];
  const float* W2   = (const float*)d_in[8];
  const float* Wout = (const float*)d_in[9];

  float* ws   = (float*)d_ws;
  float* kbuf = ws;
  float* vbuf = kbuf + 4096 * 256;
  float* qbuf = vbuf + 4096 * 256;
  float* hob  = qbuf + 4096 * 256;
  float* lrb  = hob + 4096 * 256;
  float* etab = lrb + 4096 * 4;
  float* alb  = etab + 4096 * 4;

  size_t lds1 = (size_t)(64 * 256 * 2) * 2 + 64 * 64 * 4 + 64 * 4;
  proj_kernel<<<768, 256, lds1, stream>>>(x, Wk, Wv, Wq, kbuf, vbuf, qbuf);

  gates_kernel<<<(4096 * 12) / 256, 256, 0, stream>>>(x, Wlr, Wmom, Wdec, lrb, etab, alb);

  size_t lds2 = (size_t)(8192 * 6 + 4096) * 4        // fp32 state + h1/dh1/r
              + (size_t)(8192 * 2 + 4096 * 2 + 8192) * 2  // bf16 staging
              + (size_t)(12 * 64 + 16) * 4;          // scalar arrays
  memory_kernel<<<16, 256, lds2, stream>>>(kbuf, vbuf, qbuf, lrb, etab, alb, W1, W2, hob);

  size_t lds3 = (size_t)(64 * 256 * 2) * 2;
  outproj_kernel<<<256, 256, lds3, stream>>>(hob, Wout, (float*)d_out);
}
